// GSoP_44341242364083
// MI455X (gfx1250) — compile-verified, matrix operand order checked by reading, not run on hardware
//
#include <hip/hip_runtime.h>
#include <hip/hip_bf16.h>

// ---------------------------------------------------------------------------
// GSoP block for MI455X (gfx1250, wave32, WMMA).
// conv1(1x1) GEMM via v_wmma_f32_16x16x32_bf16 with double-buffered LDS
// staging -> BN+ReLU+center -> batched covariance GEMM via WMMA ->
// tiny VALU tail (grouped conv, BN, conv2, BN, sigmoid) -> float4 gating.
// ---------------------------------------------------------------------------

typedef __attribute__((ext_vector_type(16))) __bf16 v16bf;
typedef __attribute__((ext_vector_type(8)))  float  v8f;

#define B_      32
#define CIN     512
#define MID     128
#define HW      3136      // 56*56
#define NTILE   112       // HW columns per conv1 workgroup (28 tiles)
#define NT_CNT  28
#define LDB     40        // LDS B-panel row stride in bf16 (80B, 16B aligned)
#define KSPLIT  7         // covariance K-split (448 cols each, 14 k-steps)
#define EPS     1e-5f

// round-to-nearest-even f32 -> bf16
static __device__ __forceinline__ __bf16 f2bf(float f) {
    unsigned u = __builtin_bit_cast(unsigned, f);
    unsigned r = (u + 0x7FFFu + ((u >> 16) & 1u)) >> 16;
    unsigned short s = (unsigned short)r;
    return __builtin_bit_cast(__bf16, s);
}

union Frag16 { uint4 u4[2]; v16bf v; };

// Load a 16x32 bf16 WMMA fragment (A layout; B uses the mirrored layout with
// lane = N column, same address pattern on a K-contiguous panel).
// base points at [row0][k0], ld = row stride in elements, rows k-contiguous.
static __device__ __forceinline__ v16bf load_frag(const __bf16* base, int ld, int lane) {
    int r  = lane & 15;          // M (or N) index
    int kb = (lane >> 4) << 3;   // K base: 0 or 8 (second chunk at +16)
    const __bf16* p = base + (size_t)r * ld + kb;
    Frag16 f;
    f.u4[0] = *reinterpret_cast<const uint4*>(p);        // K = kb .. kb+7
    f.u4[1] = *reinterpret_cast<const uint4*>(p + 16);   // K = kb+16 .. kb+23
    return f.v;
}

static __device__ __forceinline__ v8f wmma_bf16(v16bf a, v16bf b, v8f c) {
    return __builtin_amdgcn_wmma_f32_16x16x32_bf16(false, a, false, b,
                                                   (short)0, c, false, false);
}

// stage a 32(ch) x 112(pix) f32 tile of x into LDS, transposed to [p][c] bf16
static __device__ __forceinline__ void stage_tile(const float* __restrict__ xb,
                                                  int k0, __bf16* dst, int tid) {
    for (int idx = tid; idx < 32 * 56; idx += 256) {   // 7 exact iterations
        int c  = idx / 56;
        int p2 = (idx - c * 56) * 2;
        float2 v = *reinterpret_cast<const float2*>(&xb[(size_t)(k0 + c) * HW + p2]);
        dst[p2 * LDB + c]       = f2bf(v.x);
        dst[(p2 + 1) * LDB + c] = f2bf(v.y);
    }
}

// ---------------------------------------------------------------- kernels ---

__global__ void k_w1bf(const float* __restrict__ W1, __bf16* __restrict__ w1bf) {
    int i = blockIdx.x * 256 + threadIdx.x;   // 128*512 = 65536
    w1bf[i] = f2bf(W1[i]);
}

// conv1: feas[b, m, p] = sum_c W1[m,c]*x[b,c,p] + b1[m]; also per-row partial
// sum/sumsq (pre-BN statistics) into exclusive slots.
__global__ __launch_bounds__(256) void k_conv1(
    const float*  __restrict__ x,
    const __bf16* __restrict__ w1bf,
    const float*  __restrict__ b1,
    float* __restrict__ feas,
    float* __restrict__ psum, float* __restrict__ psq)
{
    __shared__ __attribute__((aligned(16))) __bf16 ldsB[2][NTILE * LDB];

    const int tile = blockIdx.x, b = blockIdx.y;
    const int p0   = tile * NTILE;
    const int tid  = threadIdx.x;
    const int w    = tid >> 5, lane = tid & 31;
    const int lo   = lane & 15, hi = lane >> 4;

    v8f acc[7];
    const v8f vzero = {0.f, 0.f, 0.f, 0.f, 0.f, 0.f, 0.f, 0.f};
    for (int i = 0; i < 7; ++i) acc[i] = vzero;

    const float* xb = x + ((size_t)b * CIN) * HW + p0;

    stage_tile(xb, 0, ldsB[0], tid);
    __syncthreads();

    #pragma unroll 1
    for (int step = 0; step < 16; ++step) {
        const int buf = step & 1;
        const int k0  = step * 32;

        if (step + 1 < 16) {
            // prefetch hint for the K-panel after next, then stage next panel
            if (step + 2 < 16)
                __builtin_prefetch(&xb[(size_t)(k0 + 64 + (tid & 31)) * HW + (tid >> 5) * 16], 0, 0);
            stage_tile(xb, k0 + 32, ldsB[buf ^ 1], tid);
        }

        v16bf afrag = load_frag(w1bf + (size_t)(w * 16) * CIN + k0, CIN, lane);
        v16bf bfr[7];
        #pragma unroll
        for (int nt = 0; nt < 7; ++nt)
            bfr[nt] = load_frag(ldsB[buf] + nt * 16 * LDB, LDB, lane);
        #pragma unroll
        for (int nt = 0; nt < 7; ++nt)
            acc[nt] = wmma_bf16(afrag, bfr[nt], acc[nt]);

        __syncthreads();
    }

    // epilogue: bias, store, per-row partial stats
    float rs[8], rq[8];
    #pragma unroll
    for (int r = 0; r < 8; ++r) { rs[r] = 0.f; rq[r] = 0.f; }

    #pragma unroll
    for (int nt = 0; nt < 7; ++nt) {
        #pragma unroll
        for (int r = 0; r < 8; ++r) {
            int m = w * 16 + r + 8 * hi;
            float v = acc[nt][r] + b1[m];
            feas[((size_t)b * MID + m) * HW + p0 + nt * 16 + lo] = v;
            rs[r] += v; rq[r] += v * v;
        }
    }
    #pragma unroll
    for (int r = 0; r < 8; ++r) {
        float s = rs[r], q = rq[r];
        for (int msk = 8; msk >= 1; msk >>= 1) {   // reduce across the 16-lane half
            s += __shfl_xor(s, msk);
            q += __shfl_xor(q, msk);
        }
        if (lo == 0) {
            int m  = w * 16 + r + 8 * hi;
            int wg = b * NT_CNT + tile;
            psum[(size_t)wg * MID + m] = s;
            psq [(size_t)wg * MID + m] = q;
        }
    }
}

// reduce partials -> per-channel BN scale/shift
__global__ void k_stats1(const float* __restrict__ psum, const float* __restrict__ psq,
                         const float* __restrict__ g1, const float* __restrict__ beta1,
                         float* __restrict__ sshift)
{
    int m = threadIdx.x;   // 128
    float s = 0.f, q = 0.f;
    for (int wg = 0; wg < B_ * NT_CNT; ++wg) {
        s += psum[(size_t)wg * MID + m];
        q += psq [(size_t)wg * MID + m];
    }
    const float inv = 1.f / (float)(B_ * HW);
    float mean = s * inv;
    float var  = q * inv - mean * mean;
    float sc   = g1[m] * rsqrtf(var + EPS);
    sshift[m]        = sc;
    sshift[MID + m]  = beta1[m] - mean * sc;
}

// BN + ReLU + per-(b,m) HW-mean centering -> bf16
__global__ __launch_bounds__(256) void k_bnrelu_center(
    const float* __restrict__ feas, const float* __restrict__ sshift,
    __bf16* __restrict__ feac)
{
    const int row = blockIdx.x;          // b*128 + m
    const int m   = row & (MID - 1);
    const float sc = sshift[m], sh = sshift[MID + m];
    const float* src = feas + (size_t)row * HW;
    __bf16*      dst = feac + (size_t)row * HW;
    const int tid = threadIdx.x;

    float vals[13];
    float s = 0.f;
    #pragma unroll
    for (int i = 0; i < 13; ++i) {
        int p = tid + i * 256;
        float v = 0.f;
        if (p < HW) { v = sc * src[p] + sh; v = v > 0.f ? v : 0.f; }
        vals[i] = v; s += v;
    }
    __shared__ float red[8];
    for (int msk = 16; msk >= 1; msk >>= 1) s += __shfl_xor(s, msk);
    if ((tid & 31) == 0) red[tid >> 5] = s;
    __syncthreads();
    if (tid < 8) {
        float t = red[tid];
        for (int msk = 4; msk >= 1; msk >>= 1) t += __shfl_xor(t, msk);
        if (tid == 0) red[0] = t;
    }
    __syncthreads();
    const float mean = red[0] * (1.f / (float)HW);
    #pragma unroll
    for (int i = 0; i < 13; ++i) {
        int p = tid + i * 256;
        if (p < HW) dst[p] = f2bf(vals[i] - mean);
    }
}

// cov_partial[b][ks] = feac[b][:, kchunk] * feac[b][:, kchunk]^T  (bf16 WMMA)
__global__ __launch_bounds__(256) void k_cov(const __bf16* __restrict__ feac,
                                             float* __restrict__ covp)
{
    const int ks = blockIdx.x, b = blockIdx.y;
    const int tid = threadIdx.x;
    const int w = tid >> 5, lane = tid & 31, lo = lane & 15, hi = lane >> 4;
    const __bf16* fb = feac + (size_t)b * MID * HW;

    v8f acc[8];
    const v8f vzero = {0.f, 0.f, 0.f, 0.f, 0.f, 0.f, 0.f, 0.f};
    for (int i = 0; i < 8; ++i) acc[i] = vzero;

    const int k0 = ks * (HW / KSPLIT);         // 448-wide chunk, 14 k-steps
    #pragma unroll 1
    for (int step = 0; step < 14; ++step) {
        int k = k0 + step * 32;
        v16bf afrag = load_frag(fb + (size_t)(w * 16) * HW + k, HW, lane);
        v16bf bfr[8];
        #pragma unroll
        for (int nt = 0; nt < 8; ++nt)
            bfr[nt] = load_frag(fb + (size_t)(nt * 16) * HW + k, HW, lane);
        #pragma unroll
        for (int nt = 0; nt < 8; ++nt)
            acc[nt] = wmma_bf16(afrag, bfr[nt], acc[nt]);
    }
    float* outp = covp + ((size_t)b * KSPLIT + ks) * MID * MID;
    #pragma unroll
    for (int nt = 0; nt < 8; ++nt)
        #pragma unroll
        for (int r = 0; r < 8; ++r)
            outp[(size_t)(w * 16 + r + 8 * hi) * MID + nt * 16 + lo] = acc[nt][r];
}

// grouped row-wise conv + BN over batch axis. One wave per out-channel,
// lane = batch index (wave32 == 32 batches, a perfect fit).
__global__ void k_rowwise(const float* __restrict__ covp, const float* __restrict__ Krw,
                          const float* __restrict__ brw, const float* __restrict__ grw,
                          const float* __restrict__ betarw, float* __restrict__ ybn)
{
    const int oc = blockIdx.x;        // 512
    const int b  = threadIdx.x;       // 32
    const int g  = oc >> 2;
    const float* kr = Krw + (size_t)oc * MID;
    float y = 0.f;
    for (int h = 0; h < MID; ++h) {
        float cv = 0.f;
        #pragma unroll
        for (int ks = 0; ks < KSPLIT; ++ks)
            cv += covp[(((size_t)b * KSPLIT + ks) * MID + g) * MID + h];
        y += kr[h] * cv;
    }
    y += brw[oc];
    float s = y, q = y * y;
    for (int msk = 16; msk >= 1; msk >>= 1) { s += __shfl_xor(s, msk); q += __shfl_xor(q, msk); }
    float mean = s * (1.f / 32.f);
    float var  = q * (1.f / 32.f) - mean * mean;
    ybn[b * 512 + oc] = grw[oc] * rsqrtf(var + EPS) * (y - mean) + betarw[oc];
}

// conv2 + BN over batch + sigmoid -> gate
__global__ void k_conv2_gate(const float* __restrict__ ybn, const float* __restrict__ W2,
                             const float* __restrict__ b2, const float* __restrict__ g2,
                             const float* __restrict__ beta2, float* __restrict__ gate)
{
    const int co = blockIdx.x;        // 512
    const int b  = threadIdx.x;       // 32
    const float* wr = W2  + (size_t)co * 512;
    const float* yr = ybn + (size_t)b  * 512;
    float z = b2[co];
    for (int i = 0; i < 512; ++i) z += wr[i] * yr[i];
    float s = z, q = z * z;
    for (int msk = 16; msk >= 1; msk >>= 1) { s += __shfl_xor(s, msk); q += __shfl_xor(q, msk); }
    float mean = s * (1.f / 32.f);
    float var  = q * (1.f / 32.f) - mean * mean;
    float zn = g2[co] * rsqrtf(var + EPS) * (z - mean) + beta2[co];
    gate[b * 512 + co] = 1.f / (1.f + __expf(-zn));
}

// out = x * gate[b, c]   (float4 stream)
__global__ __launch_bounds__(256) void k_gate(const float* __restrict__ x,
                                              const float* __restrict__ gate,
                                              float* __restrict__ out)
{
    size_t i4   = (size_t)blockIdx.x * 256 + threadIdx.x;
    size_t flat = i4 * 4;
    int bc = (int)(flat / HW);        // == b*512 + c, HW % 4 == 0
    float g = gate[bc];
    float4 v = reinterpret_cast<const float4*>(x)[i4];
    v.x *= g; v.y *= g; v.z *= g; v.w *= g;
    reinterpret_cast<float4*>(out)[i4] = v;
}

// ------------------------------------------------------------------- host ---

extern "C" void kernel_launch(void* const* d_in, const int* in_sizes, int n_in,
                              void* d_out, int out_size, void* d_ws, size_t ws_size,
                              hipStream_t stream) {
    const float* x     = (const float*)d_in[0];
    const float* W1    = (const float*)d_in[1];
    const float* b1    = (const float*)d_in[2];
    const float* g1    = (const float*)d_in[3];
    const float* beta1 = (const float*)d_in[4];
    const float* Krw   = (const float*)d_in[5];
    const float* brw   = (const float*)d_in[6];
    const float* grw   = (const float*)d_in[7];
    const float* betarw= (const float*)d_in[8];
    const float* W2    = (const float*)d_in[9];
    const float* b2    = (const float*)d_in[10];
    const float* g2    = (const float*)d_in[11];
    const float* beta2 = (const float*)d_in[12];
    float* out = (float*)d_out;

    // workspace layout (256B aligned slabs)
    char*  base = (char*)d_ws;
    size_t off  = 0;
    auto alloc = [&](size_t bytes) -> char* {
        char* p = base + off;
        off = (off + bytes + 255) & ~(size_t)255;
        return p;
    };
    __bf16* w1bf  = (__bf16*)alloc((size_t)MID * CIN * 2);               // 128 KB
    float*  feas  = (float*) alloc((size_t)B_ * MID * HW * 4);           // 51.4 MB
    __bf16* feac  = (__bf16*)alloc((size_t)B_ * MID * HW * 2);           // 25.7 MB
    float*  covp  = (float*) alloc((size_t)B_ * KSPLIT * MID * MID * 4); // 14.7 MB
    float*  psum  = (float*) alloc((size_t)B_ * NT_CNT * MID * 4);
    float*  psq   = (float*) alloc((size_t)B_ * NT_CNT * MID * 4);
    float*  sshift= (float*) alloc(2 * MID * 4);
    float*  ybn   = (float*) alloc((size_t)B_ * 512 * 4);
    float*  gate  = (float*) alloc((size_t)B_ * 512 * 4);
    (void)ws_size; (void)n_in; (void)in_sizes; (void)out_size;

    k_w1bf<<<(MID * CIN) / 256, 256, 0, stream>>>(W1, w1bf);

    dim3 gA(NT_CNT, B_);
    k_conv1<<<gA, 256, 0, stream>>>(x, w1bf, b1, feas, psum, psq);

    k_stats1<<<1, MID, 0, stream>>>(psum, psq, g1, beta1, sshift);

    k_bnrelu_center<<<B_ * MID, 256, 0, stream>>>(feas, sshift, feac);

    dim3 gC(KSPLIT, B_);
    k_cov<<<gC, 256, 0, stream>>>(feac, covp);

    k_rowwise<<<512, 32, 0, stream>>>(covp, Krw, brw, grw, betarw, ybn);

    k_conv2_gate<<<512, 32, 0, stream>>>(ybn, W2, b2, g2, beta2, gate);

    size_t n4 = (size_t)B_ * CIN * HW / 4;       // 12,845,056 -> 50176 blocks
    k_gate<<<(unsigned)(n4 / 256), 256, 0, stream>>>(x, gate, out);
}
